// TFGPT2MainLayer_5841155522674
// MI455X (gfx1250) — compile-verified, matrix operand order checked by reading, not run on hardware
//
#include <hip/hip_runtime.h>
#include <math.h>

// ---------------------------------------------------------------------------
// GPT-2 forward (B=4,S=1024,D=768,H=12,L=12,FF=3072) for gfx1250 (CDNA5).
// All GEMMs + attention use v_wmma_f32_16x16x32_bf16 (wave32 WMMA).
// bf16 tile staging uses global_load_async_to_lds_b128 (ASYNCcnt path),
// double-buffered LDS, software-pipelined inner loop.
// ---------------------------------------------------------------------------

typedef __attribute__((ext_vector_type(16))) __bf16 v16bf;
typedef __attribute__((ext_vector_type(8)))  float  v8f;

union FragB16 { v16bf v; uint4 q[2]; };

static __device__ __forceinline__ v8f wmma_bf16(const v16bf& a, const v16bf& b, v8f c) {
    return __builtin_amdgcn_wmma_f32_16x16x32_bf16(false, a, false, b, (short)0, c,
                                                   false, false);
}

// Async 16-byte global -> LDS copy (CDNA5, tracked by ASYNCcnt).
// lds_off: byte offset within the workgroup's LDS allocation (low 32 bits of a
// generic pointer to __shared__); gptr: per-lane global VA.
static __device__ __forceinline__ void async_copy16(unsigned lds_off, const void* gptr) {
    asm volatile("global_load_async_to_lds_b128 %0, %1, off"
                 :: "v"(lds_off), "v"((unsigned long long)(uintptr_t)gptr)
                 : "memory");
}
static __device__ __forceinline__ void wait_async0() {
    asm volatile("s_wait_asynccnt 0" ::: "memory");
}

// ---------------------------------------------------------------------------
// Tiled WMMA GEMM: C[M,N] = A[M,K](bf16) * B[K,N](f32->bf16) + bias (+res)
// Block tile 128x128, BK=32, 256 threads = 8 waves, each wave 32x64 (2x4 WMMA).
// Double-buffered LDS; A async-to-LDS; B f32->bf16 packed-pair stores, N-major.
// MODE 0: f32 out (+ optional f32 residual)  | 1: bf16 out | 2: bf16 out, GELU
// ---------------------------------------------------------------------------
template <int MODE>
__global__ __launch_bounds__(256) void wmma_gemm(
    const __bf16* __restrict__ A, const float* __restrict__ Bw,
    const float* __restrict__ bias, void* __restrict__ Out,
    const float* __restrict__ Res, int M, int N, int K)
{
    __shared__ __align__(16) __bf16 As[2][128 * 40];   // [m][k], pad to 40
    __shared__ __align__(16) __bf16 Bs[2][128 * 40];   // [n][k] (transposed)

    const int tid  = threadIdx.x;
    const int lane = tid & 31;
    const int wave = tid >> 5;
    const int half = lane >> 4;
    const int l16  = lane & 15;
    const int wm   = wave & 3;        // 0..3 : M direction (4*32 = 128)
    const int wn   = wave >> 2;       // 0..1 : N direction (2*64 = 128)
    const int m0   = blockIdx.y * 128;
    const int n0   = blockIdx.x * 128;

    // A staging: thread -> (row, 16-col segment), two async 16B copies
    const int arow = tid >> 1;                 // 0..127
    const int acs  = (tid & 1) * 16;           // 0 | 16
    const unsigned ldsA[2] = {
        (unsigned)(uintptr_t)&As[0][arow * 40 + acs],
        (unsigned)(uintptr_t)&As[1][arow * 40 + acs]
    };
    // B staging: thread -> (pair of K rows, 8 N columns) for packed stores
    const int bkr2 = (tid >> 4) * 2;           // 0,2,..,30
    const int bns  = (tid & 15) * 8;           // 0..120

    auto stage_a = [&](int k0, int buf) {
        const __bf16* ga = A + (size_t)(m0 + arow) * K + k0 + acs;
        async_copy16(ldsA[buf], ga);
        async_copy16(ldsA[buf] + 16, ga + 8);
    };
    auto load_b = [&](int k0, float* v) {
        const float4* s0 = (const float4*)(Bw + (size_t)(k0 + bkr2) * N + n0 + bns);
        const float4* s1 = (const float4*)(Bw + (size_t)(k0 + bkr2 + 1) * N + n0 + bns);
        const float4 a0 = s0[0], a1 = s0[1], b0 = s1[0], b1 = s1[1];
        v[0] = a0.x; v[1] = a0.y; v[2]  = a0.z; v[3]  = a0.w;
        v[4] = a1.x; v[5] = a1.y; v[6]  = a1.z; v[7]  = a1.w;
        v[8] = b0.x; v[9] = b0.y; v[10] = b0.z; v[11] = b0.w;
        v[12] = b1.x; v[13] = b1.y; v[14] = b1.z; v[15] = b1.w;
    };
    auto store_b = [&](const float* v, int buf) {
#pragma unroll
        for (int j = 0; j < 8; ++j) {
            union { unsigned u; __bf16 h[2]; } pk;
            pk.h[0] = (__bf16)v[j];          // k = bkr2   (low half)
            pk.h[1] = (__bf16)v[8 + j];      // k = bkr2+1 (high half)
            *(unsigned*)(&Bs[buf][(bns + j) * 40 + bkr2]) = pk.u;
        }
    };

    const v8f zero = {};
    v8f acc[2][4];
#pragma unroll
    for (int mt = 0; mt < 2; ++mt)
#pragma unroll
        for (int nt = 0; nt < 4; ++nt) acc[mt][nt] = zero;

    // ---- prologue: stage tile 0 into buffer 0 ----
    stage_a(0, 0);
    {
        float bv[16];
        load_b(0, bv);
        store_b(bv, 0);
    }
    wait_async0();
    __syncthreads();

    const int nk = K >> 5;
    for (int kt = 0; kt < nk; ++kt) {
        const int cur = kt & 1;
        const int nxt = cur ^ 1;
        const bool more = (kt + 1 < nk);
        float bn[16];
        if (more) {
            stage_a((kt + 1) * 32, nxt);       // async copies in flight
            load_b((kt + 1) * 32, bn);         // global loads in flight
            if (kt + 2 < nk)
                __builtin_prefetch(Bw + (size_t)((kt + 2) * 32 + bkr2) * N + n0 + bns,
                                   0, 1);
        }

        // ---- compute tile kt from buffer cur: 8 WMMAs per wave ----
        FragB16 fa[2], fb[4];
#pragma unroll
        for (int mt = 0; mt < 2; ++mt) {
            const int r = wm * 32 + mt * 16 + l16;
            fa[mt].q[0] = *(const uint4*)(&As[cur][r * 40 + 8 * half]);
            fa[mt].q[1] = *(const uint4*)(&As[cur][r * 40 + 16 + 8 * half]);
        }
#pragma unroll
        for (int nt = 0; nt < 4; ++nt) {
            const int c = wn * 64 + nt * 16 + l16;
            fb[nt].q[0] = *(const uint4*)(&Bs[cur][c * 40 + 8 * half]);
            fb[nt].q[1] = *(const uint4*)(&Bs[cur][c * 40 + 16 + 8 * half]);
        }
#pragma unroll
        for (int mt = 0; mt < 2; ++mt)
#pragma unroll
            for (int nt = 0; nt < 4; ++nt)
                acc[mt][nt] = wmma_bf16(fa[mt].v, fb[nt].v, acc[mt][nt]);

        if (more) {
            store_b(bn, nxt);                  // convert + packed LDS stores
            wait_async0();                     // tile kt+1 async copies done
        }
        __syncthreads();
    }

    // ---- epilogue ----
#pragma unroll
    for (int mt = 0; mt < 2; ++mt)
#pragma unroll
        for (int nt = 0; nt < 4; ++nt) {
#pragma unroll
            for (int r = 0; r < 8; ++r) {
                const int m = m0 + wm * 32 + mt * 16 + r + 8 * half;
                const int n = n0 + wn * 64 + nt * 16 + l16;
                const size_t idx = (size_t)m * N + n;
                float v = acc[mt][nt][r] + bias[n];
                if (MODE == 0) {
                    if (Res) v += Res[idx];
                    ((float*)Out)[idx] = v;
                } else if (MODE == 1) {
                    ((__bf16*)Out)[idx] = (__bf16)v;
                } else {
                    const float g = 0.5f * v * (1.0f + erff(v * 0.70710678118654752f));
                    ((__bf16*)Out)[idx] = (__bf16)g;
                }
            }
        }
}

// ---------------------------------------------------------------------------
// Flash attention, causal. One wave per (b, h, 16-query tile).
// QK^T: 2 WMMAs (DH=64). PV: 4 WMMAs per 32-key block. Online softmax.
// Q/K tiles staged with async global->LDS; V transposed via VGPRs.
// qkv is bf16 [B*S, 3*768] with q/k/v at col 0/768/1536 (+ h*64).
// ---------------------------------------------------------------------------
__global__ __launch_bounds__(32) void flash_attn(
    const __bf16* __restrict__ qkv, __bf16* __restrict__ outA)
{
    const int qt = blockIdx.x;          // 0..63
    const int bh = blockIdx.y;          // 0..47
    const int b  = bh / 12;
    const int hh = bh % 12;
    const int q0 = qt * 16;
    const int lane = threadIdx.x;
    const int half = lane >> 4;
    const int l16  = lane & 15;

    __shared__ __align__(16) __bf16 Qs[16 * 72];   // [q][dh]
    __shared__ __align__(16) __bf16 Ks[32 * 72];   // [key][dh]
    __shared__ __align__(16) __bf16 Vt[64 * 40];   // [dh][key]  (transposed)
    __shared__ __align__(16) __bf16 Ps[16 * 40];   // [q][key]

    const size_t rs = 3 * 768;
    const __bf16* base = qkv + (size_t)(b * 1024) * rs + hh * 64;

    // stage Q (16x64): 2 lanes per row, 32 bf16 each, async
    {
        const int r   = lane >> 1;
        const int seg = (lane & 1) * 32;
        const unsigned ldsQ = (unsigned)(uintptr_t)&Qs[r * 72 + seg];
        const __bf16* gq = base + (size_t)(q0 + r) * rs + seg;
#pragma unroll
        for (int j = 0; j < 4; ++j) async_copy16(ldsQ + 16 * j, gq + 8 * j);
    }
    const unsigned ldsK = (unsigned)(uintptr_t)&Ks[lane * 72];

    const v8f zero = {};
    float mrow[8], lrow[8];
    v8f o[4];
#pragma unroll
    for (int r = 0; r < 8; ++r) { mrow[r] = -1e30f; lrow[r] = 0.0f; }
#pragma unroll
    for (int t = 0; t < 4; ++t) o[t] = zero;

    const int nkb = q0 / 32 + 1;
    for (int kb = 0; kb < nkb; ++kb) {
        const int k0 = kb * 32;
        __syncthreads();
        // stage K (32x64) async and V^T (64x32) via VGPRs: one lane per key row
        {
            const __bf16* gk = base + 768 + (size_t)(k0 + lane) * rs;
#pragma unroll
            for (int j = 0; j < 8; ++j) async_copy16(ldsK + 16 * j, gk + 8 * j);

            const uint4* vsrc = (const uint4*)(base + 1536 + (size_t)(k0 + lane) * rs);
            uint4 vv[8];
#pragma unroll
            for (int j = 0; j < 8; ++j) vv[j] = vsrc[j];
            const __bf16* vb = (const __bf16*)vv;
#pragma unroll
            for (int d = 0; d < 64; ++d) Vt[d * 40 + lane] = vb[d];
        }
        wait_async0();
        __syncthreads();

        // scores: S[16 x 32] = Q(16x64) * K^T
        v8f s[2];
#pragma unroll
        for (int nt = 0; nt < 2; ++nt) {
            s[nt] = zero;
#pragma unroll
            for (int kk = 0; kk < 2; ++kk) {
                FragB16 fq, fk;
                fq.q[0] = *(const uint4*)(&Qs[l16 * 72 + kk * 32 + 8 * half]);
                fq.q[1] = *(const uint4*)(&Qs[l16 * 72 + kk * 32 + 16 + 8 * half]);
                const int kr = nt * 16 + l16;
                fk.q[0] = *(const uint4*)(&Ks[kr * 72 + kk * 32 + 8 * half]);
                fk.q[1] = *(const uint4*)(&Ks[kr * 72 + kk * 32 + 16 + 8 * half]);
                s[nt] = wmma_bf16(fq.v, fk.v, s[nt]);
            }
        }

        // online softmax over this 32-key block
        float p0[8], p1[8], sc[8];
#pragma unroll
        for (int r = 0; r < 8; ++r) {
            const int qi = q0 + r + 8 * half;
            float v0 = s[0][r] * 0.125f;                 // 1/sqrt(64)
            float v1 = s[1][r] * 0.125f;
            if (k0 + l16 > qi)       v0 = -10000.0f;     // GPT-2 mask value
            if (k0 + 16 + l16 > qi)  v1 = -10000.0f;
            float rm = fmaxf(v0, v1);
            for (int off = 1; off < 16; off <<= 1)
                rm = fmaxf(rm, __shfl_xor(rm, off, 32));
            const float nm = fmaxf(mrow[r], rm);
            sc[r] = __expf(mrow[r] - nm);
            p0[r] = __expf(v0 - nm);
            p1[r] = __expf(v1 - nm);
            float rsum = p0[r] + p1[r];
            for (int off = 1; off < 16; off <<= 1)
                rsum += __shfl_xor(rsum, off, 32);
            lrow[r] = lrow[r] * sc[r] + rsum;
            mrow[r] = nm;
        }

        // store P (bf16) to LDS
#pragma unroll
        for (int r = 0; r < 8; ++r) {
            Ps[(r + 8 * half) * 40 + l16]      = (__bf16)p0[r];
            Ps[(r + 8 * half) * 40 + 16 + l16] = (__bf16)p1[r];
        }
        __syncthreads();

        // O = O*sc + P(16x32) * V(32x64)
        FragB16 fp;
        fp.q[0] = *(const uint4*)(&Ps[l16 * 40 + 8 * half]);
        fp.q[1] = *(const uint4*)(&Ps[l16 * 40 + 16 + 8 * half]);
#pragma unroll
        for (int t = 0; t < 4; ++t) {
#pragma unroll
            for (int r = 0; r < 8; ++r) o[t][r] *= sc[r];
            FragB16 fv;
            fv.q[0] = *(const uint4*)(&Vt[(t * 16 + l16) * 40 + 8 * half]);
            fv.q[1] = *(const uint4*)(&Vt[(t * 16 + l16) * 40 + 16 + 8 * half]);
            o[t] = wmma_bf16(fp.v, fv.v, o[t]);
        }
    }

    // normalize + write bf16 merged-head output [B*S, 768]
#pragma unroll
    for (int t = 0; t < 4; ++t)
#pragma unroll
        for (int r = 0; r < 8; ++r) {
            const int m = q0 + r + 8 * half;
            const float val = o[t][r] / lrow[r];
            outA[(size_t)(b * 1024 + m) * 768 + hh * 64 + t * 16 + l16] = (__bf16)val;
        }
}

// ---------------------------------------------------------------------------
// Embedding: h[row] = wte[ids[row]] + wpe[row % S]
// ---------------------------------------------------------------------------
__global__ __launch_bounds__(256) void embed_kernel(
    const int* __restrict__ ids, const float* __restrict__ wte,
    const float* __restrict__ wpe, float* __restrict__ h)
{
    const int row = blockIdx.x;
    const int s   = row & 1023;
    const int id  = ids[row];
    const float* te = wte + (size_t)id * 768;
    const float* pe = wpe + (size_t)s * 768;
    float* hr = h + (size_t)row * 768;
    for (int i = threadIdx.x; i < 768; i += 256) hr[i] = te[i] + pe[i];
}

// ---------------------------------------------------------------------------
// LayerNorm -> bf16 activations
// ---------------------------------------------------------------------------
__global__ __launch_bounds__(256) void ln_to_bf16(
    const float* __restrict__ x, const float* __restrict__ g,
    const float* __restrict__ bta, __bf16* __restrict__ out)
{
    const int row = blockIdx.x;
    const int tid = threadIdx.x;
    __shared__ float red[256];
    const float* xr = x + (size_t)row * 768;

    float s = 0.0f;
    for (int i = tid; i < 768; i += 256) s += xr[i];
    red[tid] = s; __syncthreads();
    for (int st = 128; st > 0; st >>= 1) {
        if (tid < st) red[tid] += red[tid + st];
        __syncthreads();
    }
    const float mean = red[0] * (1.0f / 768.0f);
    __syncthreads();

    float vs = 0.0f;
    for (int i = tid; i < 768; i += 256) { const float d = xr[i] - mean; vs += d * d; }
    red[tid] = vs; __syncthreads();
    for (int st = 128; st > 0; st >>= 1) {
        if (tid < st) red[tid] += red[tid + st];
        __syncthreads();
    }
    const float rstd = rsqrtf(red[0] * (1.0f / 768.0f) + 1e-5f);

    for (int i = tid; i < 768; i += 256)
        out[(size_t)row * 768 + i] = (__bf16)((xr[i] - mean) * rstd * g[i] + bta[i]);
}

// ---------------------------------------------------------------------------
// Final: LN -> clip(0,1) -> dot with img_w[:,0]  (N=1, so plain reduction)
// ---------------------------------------------------------------------------
__global__ __launch_bounds__(256) void final_kernel(
    const float* __restrict__ x, const float* __restrict__ g,
    const float* __restrict__ bta, const float* __restrict__ w,
    float* __restrict__ out)
{
    const int row = blockIdx.x;
    const int tid = threadIdx.x;
    __shared__ float red[256];
    const float* xr = x + (size_t)row * 768;

    float s = 0.0f;
    for (int i = tid; i < 768; i += 256) s += xr[i];
    red[tid] = s; __syncthreads();
    for (int st = 128; st > 0; st >>= 1) {
        if (tid < st) red[tid] += red[tid + st];
        __syncthreads();
    }
    const float mean = red[0] * (1.0f / 768.0f);
    __syncthreads();

    float vs = 0.0f;
    for (int i = tid; i < 768; i += 256) { const float d = xr[i] - mean; vs += d * d; }
    red[tid] = vs; __syncthreads();
    for (int st = 128; st > 0; st >>= 1) {
        if (tid < st) red[tid] += red[tid + st];
        __syncthreads();
    }
    const float rstd = rsqrtf(red[0] * (1.0f / 768.0f) + 1e-5f);
    __syncthreads();

    float p = 0.0f;
    for (int i = tid; i < 768; i += 256) {
        float y = (xr[i] - mean) * rstd * g[i] + bta[i];
        y = fminf(fmaxf(y, 0.0f), 1.0f);
        p += y * w[i];
    }
    red[tid] = p; __syncthreads();
    for (int st = 128; st > 0; st >>= 1) {
        if (tid < st) red[tid] += red[tid + st];
        __syncthreads();
    }
    if (tid == 0) out[row] = red[0];
}

// ---------------------------------------------------------------------------
// Host launcher
// ---------------------------------------------------------------------------
extern "C" void kernel_launch(void* const* d_in, const int* in_sizes, int n_in,
                              void* d_out, int out_size, void* d_ws, size_t ws_size,
                              hipStream_t stream)
{
    (void)in_sizes; (void)n_in; (void)out_size; (void)ws_size;
    const int*   ids    = (const int*)d_in[0];
    const float* wte    = (const float*)d_in[1];
    const float* wpe    = (const float*)d_in[2];
    const float* ln1_g  = (const float*)d_in[3];
    const float* ln1_b  = (const float*)d_in[4];
    const float* attn_w = (const float*)d_in[5];
    const float* attn_b = (const float*)d_in[6];
    const float* proj_w = (const float*)d_in[7];
    const float* proj_b = (const float*)d_in[8];
    const float* ln2_g  = (const float*)d_in[9];
    const float* ln2_b  = (const float*)d_in[10];
    const float* fc_w   = (const float*)d_in[11];
    const float* fc_b   = (const float*)d_in[12];
    const float* mp_w   = (const float*)d_in[13];
    const float* mp_b   = (const float*)d_in[14];
    const float* lnf_g  = (const float*)d_in[15];
    const float* lnf_b  = (const float*)d_in[16];
    const float* img_w  = (const float*)d_in[17];
    float* out = (float*)d_out;

    const int R = 4096;   // B*S rows
    char* ws = (char*)d_ws;
    size_t off = 0;
    float*  h    = (float*)(ws + off);  off += (size_t)R * 768 * 4;    // residual
    __bf16* aLN  = (__bf16*)(ws + off); off += (size_t)R * 768 * 2;    // LN out
    __bf16* aATT = (__bf16*)(ws + off); off += (size_t)R * 768 * 2;    // attn out
    __bf16* aFF  = (__bf16*)(ws + off); off += (size_t)R * 3072 * 2;   // gelu out
    __bf16* qkvb = (__bf16*)(ws + off); off += (size_t)R * 2304 * 2;   // qkv

    embed_kernel<<<R, 256, 0, stream>>>(ids, wte, wpe, h);

    for (int l = 0; l < 12; ++l) {
        ln_to_bf16<<<R, 256, 0, stream>>>(h, ln1_g + l * 768, ln1_b + l * 768, aLN);
        wmma_gemm<1><<<dim3(2304 / 128, R / 128), 256, 0, stream>>>(
            aLN, attn_w + (size_t)l * 768 * 2304, attn_b + l * 2304,
            (void*)qkvb, nullptr, R, 2304, 768);
        flash_attn<<<dim3(64, 48), 32, 0, stream>>>(qkvb, aATT);
        wmma_gemm<0><<<dim3(768 / 128, R / 128), 256, 0, stream>>>(
            aATT, proj_w + (size_t)l * 768 * 768, proj_b + l * 768,
            (void*)h, h, R, 768, 768);
        ln_to_bf16<<<R, 256, 0, stream>>>(h, ln2_g + l * 768, ln2_b + l * 768, aLN);
        wmma_gemm<2><<<dim3(3072 / 128, R / 128), 256, 0, stream>>>(
            aLN, fc_w + (size_t)l * 768 * 3072, fc_b + l * 3072,
            (void*)aFF, nullptr, R, 3072, 768);
        wmma_gemm<0><<<dim3(768 / 128, R / 128), 256, 0, stream>>>(
            aFF, mp_w + (size_t)l * 3072 * 768, mp_b + l * 768,
            (void*)h, h, R, 768, 3072);
    }

    final_kernel<<<R, 256, 0, stream>>>(h, lnf_g, lnf_b, img_w, out);
}
